// CosineLayer_46729244180934
// MI455X (gfx1250) — matrix operand order repaired and checked by reading, not compiled, not run on hardware
//
#include <hip/hip_runtime.h>
#include <hip/hip_bf16.h>

typedef __attribute__((ext_vector_type(16))) __bf16 v16bf;
typedef __attribute__((ext_vector_type(8)))  float  v8f;

static constexpr int Mdim = 8192, Ndim = 8192, Kdim = 4096;
static constexpr int BM = 128, BN = 128, BK = 32;
static constexpr int LDSS = BK + 8;   // padded LDS row stride (ushorts): 80B, 16B-aligned, bank-skewed
#define COS_EPS 1e-8f

// ---------------- row inverse norms -------------------------------------
__global__ __launch_bounds__(256) void cos_norms_kernel(
    const float* __restrict__ x, const float* __restrict__ w,
    float* __restrict__ invx, float* __restrict__ invw) {
  __shared__ float red[256];
  const int row = blockIdx.x;
  const float* src;
  float* dst;
  if (row < Mdim) { src = x + (size_t)row * Kdim;          dst = invx + row; }
  else            { src = w + (size_t)(row - Mdim) * Kdim; dst = invw + (row - Mdim); }
  float s = 0.f;
  for (int i = threadIdx.x * 4; i < Kdim; i += 256 * 4) {
    float4 v = *(const float4*)(src + i);
    s += v.x * v.x + v.y * v.y + v.z * v.z + v.w * v.w;
  }
  red[threadIdx.x] = s;
  __syncthreads();
  for (int off = 128; off > 0; off >>= 1) {
    if ((int)threadIdx.x < off) red[threadIdx.x] += red[threadIdx.x + off];
    __syncthreads();
  }
  if (threadIdx.x == 0) *dst = 1.0f / fmaxf(sqrtf(red[0]), COS_EPS);
}

// ---------------- helpers -----------------------------------------------
// f32 -> bf16 pair, round-to-nearest, packed in ONE v_perm_b32:
//   result bytes {b3,b2,a3,a2}  ==  bf16(b)<<16 | bf16(a)
__device__ __forceinline__ unsigned int pk2bf(float a, float b) {
  unsigned ua = __float_as_uint(a) + 0x8000u;
  unsigned ub = __float_as_uint(b) + 0x8000u;
  return __builtin_amdgcn_perm(ub, ua, 0x07060302u);  // sel 0-3 -> ua bytes, 4-7 -> ub bytes
}
__device__ __forceinline__ uint4 cvt8(float4 a, float4 b) {
  uint4 r;
  r.x = pk2bf(a.x, a.y); r.y = pk2bf(a.z, a.w);
  r.z = pk2bf(b.x, b.y); r.w = pk2bf(b.z, b.w);
  return r;
}

// ---------------- GEMM + scale ------------------------------------------
__global__ __launch_bounds__(256) void cos_gemm_kernel(
    const float* __restrict__ X, const float* __restrict__ W,
    const float* __restrict__ invx, const float* __restrict__ invw,
    float* __restrict__ out) {
  __shared__ alignas(16) unsigned short lds[2][2][BM * LDSS];  // [buf][A=0/B=1]

  const int tid   = threadIdx.x;
  const int lane  = tid & 31;
  const int wid   = tid >> 5;
  const int waveM = wid & 3;        // 4 waves along M (32 rows each)
  const int waveN = wid >> 2;       // 2 waves along N (64 cols each)
  const int bm = blockIdx.y * BM;
  const int bn = blockIdx.x * BN;

  // global->LDS staging mapping: thread covers 16 f32 of one row per matrix
  const int lrow = tid >> 1;
  const int lcol = (tid & 1) * 16;
  const float* gA = X + (size_t)(bm + lrow) * Kdim + lcol;
  const float* gB = W + (size_t)(bn + lrow) * Kdim + lcol;
  const int sIdx = lrow * LDSS + lcol;

  v8f acc[2][4];
#pragma unroll
  for (int mt = 0; mt < 2; ++mt)
#pragma unroll
    for (int nt = 0; nt < 4; ++nt) acc[mt][nt] = (v8f)0.0f;

  // prologue: tile 0 -> buf 0
  {
    float4 a0 = *(const float4*)(gA +  0), a1 = *(const float4*)(gA +  4);
    float4 a2 = *(const float4*)(gA +  8), a3 = *(const float4*)(gA + 12);
    float4 b0 = *(const float4*)(gB +  0), b1 = *(const float4*)(gB +  4);
    float4 b2 = *(const float4*)(gB +  8), b3 = *(const float4*)(gB + 12);
    *(uint4*)(&lds[0][0][sIdx])     = cvt8(a0, a1);
    *(uint4*)(&lds[0][0][sIdx + 8]) = cvt8(a2, a3);
    *(uint4*)(&lds[0][1][sIdx])     = cvt8(b0, b1);
    *(uint4*)(&lds[0][1][sIdx + 8]) = cvt8(b2, b3);
  }
  __syncthreads();

  const int kbaseA = (lane >> 4) << 3;           // 0 / 8
  const int kbaseB = (lane >> 4) << 3;           // 0 / 8
  const int mrow = waveM * 32 + (lane & 15);
  const int nrow = waveN * 64 + (lane & 15);
  const int NT = Kdim / BK;                      // 128 K-steps

  for (int t = 0; t < NT; ++t) {
    const int cur = t & 1, nxt = cur ^ 1;

    float4 a0, a1, a2, a3, b0, b1, b2, b3;
    if (t + 1 < NT) {                            // fetch next tile into regs
      const float* pa = gA + (size_t)(t + 1) * BK;
      const float* pb = gB + (size_t)(t + 1) * BK;
      a0 = *(const float4*)(pa);      a1 = *(const float4*)(pa + 4);
      a2 = *(const float4*)(pa + 8);  a3 = *(const float4*)(pa + 12);
      b0 = *(const float4*)(pb);      b1 = *(const float4*)(pb + 4);
      b2 = *(const float4*)(pb + 8);  b3 = *(const float4*)(pb + 12);
      __builtin_prefetch(pa + 2 * BK, 0, 0);     // global_prefetch_b8, 2 tiles ahead
      __builtin_prefetch(pb + 2 * BK, 0, 0);
    }

    // build fragments from LDS (ds_load_b128) and run 8 WMMAs
    union { v16bf v; uint4 q[2]; } afrag[2], bfrag[4];
    const unsigned short* sA = lds[cur][0];
    const unsigned short* sB = lds[cur][1];
#pragma unroll
    for (int mt = 0; mt < 2; ++mt) {
      const unsigned short* p = sA + (mrow + mt * 16) * LDSS;
      afrag[mt].q[0] = *(const uint4*)(p + kbaseA);        // K 0..7  | 8..15
      afrag[mt].q[1] = *(const uint4*)(p + 16 + kbaseA);   // K 16..23|24..31
    }
#pragma unroll
    for (int nt = 0; nt < 4; ++nt) {
      const unsigned short* p = sB + (nrow + nt * 16) * LDSS + kbaseB;
      bfrag[nt].q[0] = *(const uint4*)(p);                 // K 0..7  | 8..15
      bfrag[nt].q[1] = *(const uint4*)(p + 16);            // K 16..23|24..31
    }
#pragma unroll
    for (int mt = 0; mt < 2; ++mt)
#pragma unroll
      for (int nt = 0; nt < 4; ++nt)
        acc[mt][nt] = __builtin_amdgcn_wmma_f32_16x16x32_bf16(
            false, afrag[mt].v, false, bfrag[nt].v,
            (short)0, acc[mt][nt], false, false);

    if (t + 1 < NT) {                            // stage next tile into other buffer
      *(uint4*)(&lds[nxt][0][sIdx])     = cvt8(a0, a1);
      *(uint4*)(&lds[nxt][0][sIdx + 8]) = cvt8(a2, a3);
      *(uint4*)(&lds[nxt][1][sIdx])     = cvt8(b0, b1);
      *(uint4*)(&lds[nxt][1][sIdx + 8]) = cvt8(b2, b3);
      __syncthreads();
    }
  }

  // epilogue: out = dot * invx[row] * invw[col]
  const int colbase = bn + waveN * 64 + (lane & 15);
  const int rowhalf = (lane >> 4) << 3;          // 0 / 8
#pragma unroll
  for (int mt = 0; mt < 2; ++mt) {
    const int row0 = bm + waveM * 32 + mt * 16 + rowhalf;
    float4 ix0 = *(const float4*)(invx + row0);
    float4 ix1 = *(const float4*)(invx + row0 + 4);
    float ixv[8] = {ix0.x, ix0.y, ix0.z, ix0.w, ix1.x, ix1.y, ix1.z, ix1.w};
#pragma unroll
    for (int nt = 0; nt < 4; ++nt) {
      const int col = colbase + nt * 16;
      const float iw = invw[col];
#pragma unroll
      for (int r = 0; r < 8; ++r)
        out[(size_t)(row0 + r) * Ndim + col] = acc[mt][nt][r] * ixv[r] * iw;
    }
  }
}

// ---------------- launch -------------------------------------------------
extern "C" void kernel_launch(void* const* d_in, const int* in_sizes, int n_in,
                              void* d_out, int out_size, void* d_ws, size_t ws_size,
                              hipStream_t stream) {
  (void)in_sizes; (void)n_in; (void)out_size; (void)ws_size;
  const float* x = (const float*)d_in[0];
  const float* w = (const float*)d_in[1];
  float* out  = (float*)d_out;
  float* invx = (float*)d_ws;          // 8192 f32
  float* invw = invx + Mdim;           // 8192 f32  (64 KB total)

  cos_norms_kernel<<<Mdim + Ndim, 256, 0, stream>>>(x, w, invx, invw);

  dim3 grid(Ndim / BN, Mdim / BM);     // 64 x 64 blocks
  cos_gemm_kernel<<<grid, 256, 0, stream>>>(x, w, invx, invw, out);
}
